// GNE_3418793968191
// MI455X (gfx1250) — matrix operand order repaired and checked by reading, not compiled, not run on hardware
//
#include <hip/hip_runtime.h>

typedef __attribute__((ext_vector_type(16))) _Float16 v16h;
typedef __attribute__((ext_vector_type(4)))  _Float16 h4;
typedef __attribute__((ext_vector_type(2)))  _Float16 h2;
typedef __attribute__((ext_vector_type(8)))  float    v8f;

#define BB   4
#define NN   256
#define TT   2048
#define HTD  64
#define LL   4
#define BNT  (BB * NN * TT)   // 2,097,152
#define BN   (BB * NN)        // 1,024
#define WREP (LL * 3 * NN * NN) // 786,432 repacked conv weights

#define LDT  48               // LDS row stride in halves (96B; 32B-aligned fragments, non-2^n banks)

// ---------------------------------------------------------------------------
// WMMA 16x16x32 f16->f32 fragment-order LDS layout (wave32).
// A (16Mx32K): lane fragment = Ash[row*LDT + kh*16 .. +16) with K kept contiguous at
//   staging (permk unused) so element j maps to k = kh*16 + j.
// B (32Kx16N): stored transposed [col][k]; lane fragment = Bsh[col*LDT+kh*16..)
//   matching element j -> k = kh*16 + j.
// C/D f32: acc[r] -> M = r + 8*kh, N = lane&15.
// ---------------------------------------------------------------------------
__device__ __forceinline__ int permk(int k) {
  return ((k >> 3) & 1) * 16 + ((k >> 4) & 1) * 8 + (k & 7);
}

#define WMMA(D, A, Bf) \
  D = __builtin_amdgcn_wmma_f32_16x16x32_f16(false, A, false, Bf, (short)0, D, false, false)

#define FRAG(BASE, ROW) (*(const v16h*)&(BASE)[(ROW) * LDT + kh * 16])

// Stage 32x32 A tile from row-major f32 source (row r at src + r*ld + kc).
// float4 global load -> h4 ds_store_b64 (k kept contiguous, no permutation).
__device__ __forceinline__ void stage_A32(const float* __restrict__ src, int ld,
                                          int kc, int tid, _Float16* Ash) {
#pragma unroll
  for (int s = 0; s < 2; ++s) {
    int qi = tid + 128 * s;
    int r = qi >> 3, k0 = (qi & 7) * 4;
    float4 v = *(const float4*)(src + (size_t)r * ld + kc + k0);
    h4 p = {(_Float16)v.x, (_Float16)v.y, (_Float16)v.z, (_Float16)v.w};
    *(h4*)(Ash + r * LDT + k0) = p;
  }
}

// Stage 32x32 B tile, act-style: B[k][c] = act[(kc+k)*TT + t0 + c].
// Two float4 row loads paired across k -> h2 (b32) transposed stores.
__device__ __forceinline__ void stage_B_act(const float* __restrict__ act,
                                            int kc, int t0, int lane, _Float16* Bsh) {
  const int c4 = (lane & 7) * 4;
  const int kb = (lane >> 3) * 2;
#pragma unroll
  for (int s = 0; s < 4; ++s) {
    int kp = s * 8 + kb;
    const float* p0 = act + (size_t)(kc + kp) * TT + t0 + c4;
    float4 r0 = *(const float4*)p0;
    float4 r1 = *(const float4*)(p0 + TT);
    h2 q0 = {(_Float16)r0.x, (_Float16)r1.x};
    h2 q1 = {(_Float16)r0.y, (_Float16)r1.y};
    h2 q2 = {(_Float16)r0.z, (_Float16)r1.z};
    h2 q3 = {(_Float16)r0.w, (_Float16)r1.w};
    *(h2*)(Bsh + (c4 + 0) * LDT + kp) = q0;
    *(h2*)(Bsh + (c4 + 1) * LDT + kp) = q1;
    *(h2*)(Bsh + (c4 + 2) * LDT + kp) = q2;
    *(h2*)(Bsh + (c4 + 3) * LDT + kp) = q3;
  }
}

// Stage 48x32 wide B tile for conv: col cw (0..47) <-> global t = tw0 + cw,
// zero-filled where t < 0 (tw0 is 16-aligned so each float4 quad is all-or-none).
__device__ __forceinline__ void stage_B_wide(const float* __restrict__ act,
                                             int kc, int tw0, int lane, _Float16* Bsh) {
  const int c4 = (lane & 3) * 4;       // quad within 16-col subtile
  const int kb = (lane >> 2) * 2;      // paired k rows: 0,2,...,14
#pragma unroll
  for (int sub = 0; sub < 3; ++sub) {
    const int cb = sub * 16;
    const int tq = tw0 + cb + c4;
    const bool ok = (tq >= 0);
#pragma unroll
    for (int s = 0; s < 2; ++s) {
      int kp = s * 16 + kb;
      const float* p0 = act + (size_t)(kc + kp) * TT + tq;
      float4 r0 = {0.f, 0.f, 0.f, 0.f}, r1 = {0.f, 0.f, 0.f, 0.f};
      if (ok) { r0 = *(const float4*)p0; r1 = *(const float4*)(p0 + TT); }
      h2 q0 = {(_Float16)r0.x, (_Float16)r1.x};
      h2 q1 = {(_Float16)r0.y, (_Float16)r1.y};
      h2 q2 = {(_Float16)r0.z, (_Float16)r1.z};
      h2 q3 = {(_Float16)r0.w, (_Float16)r1.w};
      *(h2*)(Bsh + (cb + c4 + 0) * LDT + kp) = q0;
      *(h2*)(Bsh + (cb + c4 + 1) * LDT + kp) = q1;
      *(h2*)(Bsh + (cb + c4 + 2) * LDT + kp) = q2;
      *(h2*)(Bsh + (cb + c4 + 3) * LDT + kp) = q3;
    }
  }
}

// Stage 32x32 B tile, W-style (gath): B[k][c] = W[(u0+c)*TT + kc+k].
// K-contiguous rows: float4 along k -> one h4 ds_store_b64.
__device__ __forceinline__ void stage_B_w(const float* __restrict__ W,
                                          int kc, int u0, int lane, _Float16* Bsh) {
  const int k4 = (lane & 7) * 4;
  const int cb = lane >> 3;
#pragma unroll
  for (int s = 0; s < 8; ++s) {
    int c = s * 4 + cb;
    float4 v = *(const float4*)(W + (size_t)(u0 + c) * TT + kc + k4);
    h4 p = {(_Float16)v.x, (_Float16)v.y, (_Float16)v.z, (_Float16)v.w};
    *(h4*)(Bsh + c * LDT + k4) = p;
  }
}

// ---------------------------------------------------------------------------
// 0) Repack conv weights: cw[l][m][i][tap] -> wt[l][tap][m][i] (row-contiguous)
// ---------------------------------------------------------------------------
__global__ __launch_bounds__(256) void repack_w_kernel(
    const float* __restrict__ cw, float* __restrict__ wt) {
  int idx = blockIdx.x * 256 + threadIdx.x;
  if (idx < WREP) {
    int i   = idx % NN;
    int m   = (idx / NN) % NN;
    int tap = (idx / (NN * NN)) % 3;
    int l   = idx / (3 * NN * NN);
    wt[idx] = cw[(((size_t)l * NN + m) * NN + i) * 3 + tap];
  }
}

// ---------------------------------------------------------------------------
// 1) Causal dilated conv (kernel 3, dilation d) + bias + ReLU
//    z[b,m,t] = relu(cb[m] + sum_{tap,i} wrep[tap][m][i] * yin[b,i,t-(2-tap)*d])
//    One wide 48-col B tile per K chunk serves all 3 taps via row offsets.
// ---------------------------------------------------------------------------
__global__ __launch_bounds__(128) void conv_kernel(
    const float* __restrict__ yin, const float* __restrict__ wrep,
    const float* __restrict__ cb, float* __restrict__ z, int dil) {
  __shared__ alignas(32) _Float16 Ash[3][32 * LDT];
  __shared__ alignas(32) _Float16 Bsh[4][48 * LDT];
  const int tid  = threadIdx.x;
  const int wave = tid >> 5;
  const int lane = tid & 31;
  const int kh   = lane >> 4;
  const int rc   = lane & 15;
  const int b    = blockIdx.z;
  const int m0   = blockIdx.y * 32;
  const int t0   = blockIdx.x * 128 + wave * 32;
  const int tw0  = t0 - 16;                     // wide-tile time base (16-aligned)
  const float* yin_b = yin + (size_t)b * NN * TT;

  v8f acc00 = {}, acc01 = {}, acc10 = {}, acc11 = {};
  for (int kc = 0; kc < NN; kc += 32) {
    __syncthreads();
#pragma unroll
    for (int tap = 0; tap < 3; ++tap)
      stage_A32(wrep + ((size_t)tap * NN + m0) * NN, NN, kc, tid, &Ash[tap][0]);
    stage_B_wide(yin_b, kc, tw0, lane, &Bsh[wave][0]);
    __syncthreads();
#pragma unroll
    for (int tap = 0; tap < 3; ++tap) {
      const int off = 16 - (2 - tap) * dil;     // wide-tile row offset for this tap
      v16h a0 = FRAG(Ash[tap], rc), a1 = FRAG(Ash[tap], 16 + rc);
      v16h b0 = FRAG(Bsh[wave], rc + off), b1 = FRAG(Bsh[wave], 16 + rc + off);
      WMMA(acc00, a0, b0); WMMA(acc01, a0, b1);
      WMMA(acc10, a1, b0); WMMA(acc11, a1, b1);
    }
  }
  v8f accs[4] = {acc00, acc01, acc10, acc11};
#pragma unroll
  for (int q = 0; q < 4; ++q) {
    int mi = q >> 1, ci = q & 1;
#pragma unroll
    for (int r = 0; r < 8; ++r) {
      int m = m0 + mi * 16 + r + 8 * kh;
      int t = t0 + ci * 16 + rc;
      float v = accs[q][r] + cb[m];
      z[(b * NN + m) * TT + t] = v > 0.0f ? v : 0.0f;
    }
  }
}

// ---------------------------------------------------------------------------
// 2) 1x1 projection + bias + residual:  yout = yin + pb + pw @ z
// ---------------------------------------------------------------------------
__global__ __launch_bounds__(128) void proj_kernel(
    const float* __restrict__ yin, const float* __restrict__ z,
    const float* __restrict__ pw, const float* __restrict__ pb,
    float* __restrict__ yout) {
  __shared__ alignas(32) _Float16 Ash[32 * LDT];
  __shared__ alignas(32) _Float16 Bsh[4][32 * LDT];
  const int tid = threadIdx.x, wave = tid >> 5, lane = tid & 31;
  const int kh = lane >> 4, rc = lane & 15;
  const int b = blockIdx.z, m0 = blockIdx.y * 32;
  const int t0 = blockIdx.x * 128 + wave * 32;
  const float* z_b = z + (size_t)b * NN * TT;

  v8f acc00 = {}, acc01 = {}, acc10 = {}, acc11 = {};
  for (int kc = 0; kc < NN; kc += 32) {
    __syncthreads();
    stage_A32(pw + (size_t)m0 * NN, NN, kc, tid, Ash);
    stage_B_act(z_b, kc, t0, lane, &Bsh[wave][0]);
    __syncthreads();
    v16h a0 = FRAG(Ash, rc), a1 = FRAG(Ash, 16 + rc);
    v16h b0 = FRAG(Bsh[wave], rc), b1 = FRAG(Bsh[wave], 16 + rc);
    WMMA(acc00, a0, b0); WMMA(acc01, a0, b1);
    WMMA(acc10, a1, b0); WMMA(acc11, a1, b1);
  }
  v8f accs[4] = {acc00, acc01, acc10, acc11};
#pragma unroll
  for (int q = 0; q < 4; ++q) {
    int mi = q >> 1, ci = q & 1;
#pragma unroll
    for (int r = 0; r < 8; ++r) {
      int m = m0 + mi * 16 + r + 8 * kh;
      int idx = (b * NN + m) * TT + (t0 + ci * 16 + rc);
      yout[idx] = yin[idx] + pb[m] + accs[q][r];
    }
  }
}

// ---------------------------------------------------------------------------
// 3) GAT projection:  H[b,n,u] = sum_t y[b,n,t] * W[u,t]   (K = T = 2048)
// ---------------------------------------------------------------------------
__global__ __launch_bounds__(128) void gath_kernel(
    const float* __restrict__ y, const float* __restrict__ W,
    float* __restrict__ H) {
  __shared__ alignas(32) _Float16 Ash[32 * LDT];
  __shared__ alignas(32) _Float16 Bsh[4][32 * LDT];
  const int tid = threadIdx.x, wave = tid >> 5, lane = tid & 31;
  const int kh = lane >> 4, rc = lane & 15;
  const int b = blockIdx.z, n0 = blockIdx.y * 32;
  const int u0 = blockIdx.x * 128 + wave * 32;
  const float* yrows = y + (size_t)(b * NN + n0) * TT;

  v8f acc00 = {}, acc01 = {}, acc10 = {}, acc11 = {};
  for (int kc = 0; kc < TT; kc += 32) {
    if (kc + 32 < TT) {                         // prefetch next K chunk (global_prefetch_b8)
      __builtin_prefetch(&yrows[(size_t)(tid & 31) * TT + kc + 32], 0, 1);
      __builtin_prefetch(&W[(size_t)(u0 + (tid & 31)) * TT + kc + 32], 0, 1);
    }
    __syncthreads();
    stage_A32(yrows, TT, kc, tid, Ash);
    stage_B_w(W, kc, u0, lane, &Bsh[wave][0]);
    __syncthreads();
    v16h a0 = FRAG(Ash, rc), a1 = FRAG(Ash, 16 + rc);
    v16h b0 = FRAG(Bsh[wave], rc), b1 = FRAG(Bsh[wave], 16 + rc);
    WMMA(acc00, a0, b0); WMMA(acc01, a0, b1);
    WMMA(acc10, a1, b0); WMMA(acc11, a1, b1);
  }
  v8f accs[4] = {acc00, acc01, acc10, acc11};
#pragma unroll
  for (int q = 0; q < 4; ++q) {
    int mi = q >> 1, ci = q & 1;
#pragma unroll
    for (int r = 0; r < 8; ++r) {
      int n = n0 + mi * 16 + r + 8 * kh;
      H[(b * NN + n) * TT + (u0 + ci * 16 + rc)] = accs[q][r];
    }
  }
}

// ---------------------------------------------------------------------------
// 4) ei = H . a1 , ej = H . a2  (per b,n row; K=2048 reduction)
// ---------------------------------------------------------------------------
__global__ __launch_bounds__(128) void eiej_kernel(
    const float* __restrict__ H, const float* __restrict__ a1,
    const float* __restrict__ a2, float* __restrict__ ei, float* __restrict__ ej) {
  __shared__ float r1[128], r2[128];
  const int bn = blockIdx.x;
  const float* h = H + (size_t)bn * TT;
  float s1 = 0.f, s2 = 0.f;
  for (int t = threadIdx.x; t < TT; t += 128) {
    float hv = h[t];
    s1 += hv * a1[t];
    s2 += hv * a2[t];
  }
  r1[threadIdx.x] = s1;
  r2[threadIdx.x] = s2;
  __syncthreads();
  for (int s = 64; s > 0; s >>= 1) {
    if (threadIdx.x < s) {
      r1[threadIdx.x] += r1[threadIdx.x + s];
      r2[threadIdx.x] += r2[threadIdx.x + s];
    }
    __syncthreads();
  }
  if (threadIdx.x == 0) { ei[bn] = r1[0]; ej[bn] = r2[0]; }
}

// ---------------------------------------------------------------------------
// 5) cond[b,n] = ht_b[n] + sum_h ht[b,h] * ht_w[n,h]
// ---------------------------------------------------------------------------
__global__ __launch_bounds__(256) void cond_kernel(
    const float* __restrict__ ht, const float* __restrict__ htw,
    const float* __restrict__ htb, float* __restrict__ cond) {
  int idx = blockIdx.x * 256 + threadIdx.x;
  if (idx < BN) {
    int b = idx / NN, n = idx % NN;
    float s = htb[n];
#pragma unroll 8
    for (int h = 0; h < HTD; ++h) s += ht[b * HTD + h] * htw[n * HTD + h];
    cond[idx] = s;
  }
}

// ---------------------------------------------------------------------------
// 6) attention row: e = leaky(ei_i + ej_j), softmax over j -> A
// ---------------------------------------------------------------------------
__global__ __launch_bounds__(256) void softmax_kernel(
    const float* __restrict__ ei, const float* __restrict__ ej,
    float* __restrict__ Aout) {
  __shared__ float red[256];
  const int bi = blockIdx.x;           // b*N + i
  const int b = bi / NN;
  const int j = threadIdx.x;
  float e = ei[bi] + ej[b * NN + j];
  e = (e > 0.f) ? e : 0.2f * e;
  red[j] = e;
  __syncthreads();
  for (int s = 128; s > 0; s >>= 1) {
    if (j < s) red[j] = fmaxf(red[j], red[j + s]);
    __syncthreads();
  }
  float mx = red[0];
  __syncthreads();
  float ex = __expf(e - mx);
  red[j] = ex;
  __syncthreads();
  for (int s = 128; s > 0; s >>= 1) {
    if (j < s) red[j] += red[j + s];
    __syncthreads();
  }
  Aout[(size_t)bi * NN + j] = ex * (1.0f / red[0]);
}

// ---------------------------------------------------------------------------
// 7) Y[b,i,t] = sum_j A[b,i,j] * H[b,j,t] + cond[b,i]
// ---------------------------------------------------------------------------
__global__ __launch_bounds__(128) void gaty_kernel(
    const float* __restrict__ Amat, const float* __restrict__ H,
    const float* __restrict__ cond, float* __restrict__ Y) {
  __shared__ alignas(32) _Float16 Ash[32 * LDT];
  __shared__ alignas(32) _Float16 Bsh[4][32 * LDT];
  const int tid = threadIdx.x, wave = tid >> 5, lane = tid & 31;
  const int kh = lane >> 4, rc = lane & 15;
  const int b = blockIdx.z, i0 = blockIdx.y * 32;
  const int t0 = blockIdx.x * 128 + wave * 32;
  const float* H_b = H + (size_t)b * NN * TT;

  v8f acc00 = {}, acc01 = {}, acc10 = {}, acc11 = {};
  for (int kc = 0; kc < NN; kc += 32) {
    __syncthreads();
    stage_A32(Amat + (size_t)(b * NN + i0) * NN, NN, kc, tid, Ash);
    stage_B_act(H_b, kc, t0, lane, &Bsh[wave][0]);
    __syncthreads();
    v16h a0 = FRAG(Ash, rc), a1 = FRAG(Ash, 16 + rc);
    v16h b0 = FRAG(Bsh[wave], rc), b1 = FRAG(Bsh[wave], 16 + rc);
    WMMA(acc00, a0, b0); WMMA(acc01, a0, b1);
    WMMA(acc10, a1, b0); WMMA(acc11, a1, b1);
  }
  v8f accs[4] = {acc00, acc01, acc10, acc11};
#pragma unroll
  for (int q = 0; q < 4; ++q) {
    int mi = q >> 1, ci = q & 1;
#pragma unroll
    for (int r = 0; r < 8; ++r) {
      int i = i0 + mi * 16 + r + 8 * kh;
      Y[(b * NN + i) * TT + (t0 + ci * 16 + rc)] = accs[q][r] + cond[b * NN + i];
    }
  }
}

// ---------------------------------------------------------------------------
// 8) out[b,o,t] = out_b[o] + sum_i out_w[o,i] * Y[b,i,t]
// ---------------------------------------------------------------------------
__global__ __launch_bounds__(128) void out_kernel(
    const float* __restrict__ Y, const float* __restrict__ ow,
    const float* __restrict__ ob, float* __restrict__ outp) {
  __shared__ alignas(32) _Float16 Ash[32 * LDT];
  __shared__ alignas(32) _Float16 Bsh[4][32 * LDT];
  const int tid = threadIdx.x, wave = tid >> 5, lane = tid & 31;
  const int kh = lane >> 4, rc = lane & 15;
  const int b = blockIdx.z, o0 = blockIdx.y * 32;
  const int t0 = blockIdx.x * 128 + wave * 32;
  const float* Y_b = Y + (size_t)b * NN * TT;

  v8f acc00 = {}, acc01 = {}, acc10 = {}, acc11 = {};
  for (int kc = 0; kc < NN; kc += 32) {
    __syncthreads();
    stage_A32(ow + (size_t)o0 * NN, NN, kc, tid, Ash);
    stage_B_act(Y_b, kc, t0, lane, &Bsh[wave][0]);
    __syncthreads();
    v16h a0 = FRAG(Ash, rc), a1 = FRAG(Ash, 16 + rc);
    v16h b0 = FRAG(Bsh[wave], rc), b1 = FRAG(Bsh[wave], 16 + rc);
    WMMA(acc00, a0, b0); WMMA(acc01, a0, b1);
    WMMA(acc10, a1, b0); WMMA(acc11, a1, b1);
  }
  v8f accs[4] = {acc00, acc01, acc10, acc11};
#pragma unroll
  for (int q = 0; q < 4; ++q) {
    int mi = q >> 1, ci = q & 1;
#pragma unroll
    for (int r = 0; r < 8; ++r) {
      int o = o0 + mi * 16 + r + 8 * kh;
      outp[(b * NN + o) * TT + (t0 + ci * 16 + rc)] = accs[q][r] + ob[o];
    }
  }
}

// ---------------------------------------------------------------------------
extern "C" void kernel_launch(void* const* d_in, const int* in_sizes, int n_in,
                              void* d_out, int out_size, void* d_ws, size_t ws_size,
                              hipStream_t stream) {
  (void)in_sizes; (void)n_in; (void)out_size; (void)ws_size;
  const float* Xk     = (const float*)d_in[0];
  const float* ht     = (const float*)d_in[1];
  const float* conv_w = (const float*)d_in[2];
  const float* conv_b = (const float*)d_in[3];
  const float* proj_w = (const float*)d_in[4];
  const float* proj_b = (const float*)d_in[5];
  const float* gat_W  = (const float*)d_in[6];
  const float* gat_a1 = (const float*)d_in[7];
  const float* gat_a2 = (const float*)d_in[8];
  const float* out_w  = (const float*)d_in[9];
  const float* out_b  = (const float*)d_in[10];
  const float* ht_w   = (const float*)d_in[11];
  const float* ht_b   = (const float*)d_in[12];

  float* outp = (float*)d_out;            // final out, also conv-z scratch early
  float* Aout = (float*)d_out + BNT;      // attention matrix output

  float* ws   = (float*)d_ws;
  float* yA   = ws;                       // ping buffer (later: H)
  float* yB   = ws + (size_t)BNT;         // pong buffer (final y, later: Y)
  float* ei   = ws + 2 * (size_t)BNT;
  float* ej   = ei + BN;
  float* cond = ej + BN;
  float* wrep = cond + BN;                // repacked conv weights [L][tap][m][i]

  dim3 ggrid(TT / 128, NN / 32, BB);      // 16 x 8 x 4 = 512 blocks
  dim3 gblock(128);

  repack_w_kernel<<<dim3(WREP / 256), dim3(256), 0, stream>>>(conv_w, wrep);

  // residual dilated temporal stack (ping-pong through yA/yB, z in d_out)
  const float* yin = Xk;
  float* bufs[2] = { yA, yB };
  for (int l = 0; l < LL; ++l) {
    float* yout = bufs[l & 1];
    conv_kernel<<<ggrid, gblock, 0, stream>>>(yin, wrep + (size_t)l * 3 * NN * NN,
                                              conv_b + l * NN, outp, 1 << l);
    proj_kernel<<<ggrid, gblock, 0, stream>>>(yin, outp, proj_w + (size_t)l * NN * NN,
                                              proj_b + l * NN, yout);
    yin = yout;
  }
  // yin == yB (final y); yA is now free -> holds H
  float* H = yA;
  gath_kernel<<<ggrid, gblock, 0, stream>>>(yin, gat_W, H);
  eiej_kernel<<<dim3(BN), dim3(128), 0, stream>>>(H, gat_a1, gat_a2, ei, ej);
  cond_kernel<<<dim3((BN + 255) / 256), dim3(256), 0, stream>>>(ht, ht_w, ht_b, cond);
  softmax_kernel<<<dim3(BN), dim3(256), 0, stream>>>(ei, ej, Aout);
  // yB no longer needed -> holds Y
  float* Y = yB;
  gaty_kernel<<<ggrid, gblock, 0, stream>>>(Aout, H, cond, Y);
  out_kernel<<<ggrid, gblock, 0, stream>>>(Y, out_w, out_b, outp);
}
